// UniversalP_69896297775683
// MI455X (gfx1250) — compile-verified, matrix operand order checked by reading, not run on hardware
//
#include <hip/hip_runtime.h>
#include <hip/hip_bf16.h>

#define NNODES   50000
#define FEATS    128
#define CLASSES  10
#define HIDDEN   64
#define NEDGES   1600000
#define ADJ_HID  131       // hidden width of adjustment MLP
#define DEPTH    10
#define ALPHA    0.9f
#define XA_STRIDE 144      // 131 padded to 9 tiles of 16
#define NTILES   (NNODES / 16)   // 3125, exact

typedef __attribute__((ext_vector_type(16))) __bf16    bf16x16;
typedef __attribute__((ext_vector_type(8)))  float     f32x8;
typedef __attribute__((ext_vector_type(4)))  unsigned  u32x4;

// ---------------- helpers ----------------

__device__ __forceinline__ __bf16 f2bf(float f) {
  unsigned u = __builtin_bit_cast(unsigned, f);
  unsigned r = u + 0x7FFFu + ((u >> 16) & 1u);   // round-to-nearest-even
  unsigned short h = (unsigned short)(r >> 16);
  return __builtin_bit_cast(__bf16, h);
}

struct frag_bits { u32x4 lo; u32x4 hi; };        // 32 bytes

// Load a 16x32 bf16 WMMA A/B fragment from a row-major bf16 matrix.
// ISA 7.12.2: lane<16 -> row=lane,    K = {koff+0..7, koff+16..23}
//             lane>=16 -> row=lane-16, K = {koff+8..15, koff+24..31}
// Each half is 8 contiguous bf16 = one 16B load; offsets are 16B aligned.
__device__ __forceinline__ bf16x16 load_frag(const __bf16* base, int row,
                                             int stride, int koff, int lane) {
  const __bf16* p = base + (size_t)row * stride + koff + ((lane >> 4) << 3);
  frag_bits b;
  b.lo = *(const u32x4*)p;
  b.hi = *(const u32x4*)(p + 16);
  return __builtin_bit_cast(bf16x16, b);
}

__device__ __forceinline__ f32x8 wmma_bf16(bf16x16 a, bf16x16 b, f32x8 c) {
  // (neg_a, A, neg_b, B, c_mod, C, reuse_a, reuse_b)
  return __builtin_amdgcn_wmma_f32_16x16x32_bf16(false, a, false, b,
                                                 (short)0, c, false, false);
}

// ---------------- bf16 repack prepass (runs once, memory-bound) ----------------

__global__ void cvt_x_kernel(const float* __restrict__ x, __bf16* __restrict__ xb) {
  int i = blockIdx.x * blockDim.x + threadIdx.x;
  if (i < NNODES * FEATS) xb[i] = f2bf(x[i]);
}

__global__ void cvt_w_kernel(const float* __restrict__ w1, const float* __restrict__ w2,
                             const float* __restrict__ wa1,
                             __bf16* __restrict__ w1b, __bf16* __restrict__ w2b,
                             __bf16* __restrict__ wab) {
  int i = blockIdx.x * blockDim.x + threadIdx.x;
  if (i < HIDDEN * FEATS) w1b[i] = f2bf(w1[i]);          // [64,128]
  if (i < 16 * HIDDEN) {                                  // [16,64], rows>=10 zero
    int rr = i / HIDDEN, k = i % HIDDEN;
    w2b[i] = (rr < CLASSES) ? f2bf(w2[rr * HIDDEN + k]) : f2bf(0.0f);
  }
  if (i < XA_STRIDE * FEATS) {                            // [144,128] = wa1[:,11:139], rows>=131 zero
    int rr = i / FEATS, k = i % FEATS;
    wab[i] = (rr < ADJ_HID) ? f2bf(wa1[(size_t)rr * 139 + 11 + k]) : f2bf(0.0f);
  }
}

// ---------------- graph prep / diffusion kernels ----------------

__global__ void zero_deg_kernel(float* deg) {
  int i = blockIdx.x * blockDim.x + threadIdx.x;
  if (i < NNODES) deg[i] = 0.0f;
}

__global__ void count_deg_kernel(const long long* edges, float* deg) {
  int e = blockIdx.x * blockDim.x + threadIdx.x;
  if (e >= NEDGES) return;
  int s = (int)edges[e];                 // src row
  unsafeAtomicAdd(&deg[s], 1.0f);
}

__global__ void compute_r_kernel(const float* deg, float* r) {
  int i = blockIdx.x * blockDim.x + threadIdx.x;
  if (i < NNODES) r[i] = rsqrtf(deg[i] + 1.0f);
}

__global__ void edge_prep_kernel(const long long* edges, const float* r,
                                 int* src32, int* dst32, float* ew) {
  int e = blockIdx.x * blockDim.x + threadIdx.x;
  if (e >= NEDGES) return;
  int s = (int)edges[e];
  int d = (int)edges[NEDGES + e];
  src32[e] = s;
  dst32[e] = d;
  ew[e] = r[s] * r[d];
}

// agg = (r*r) * h   (self-loop term seeds the accumulator)
__global__ void init_agg_kernel(const float* r, const float* h, float* agg) {
  int i = blockIdx.x * blockDim.x + threadIdx.x;
  if (i >= NNODES * CLASSES) return;
  float rn = r[i / CLASSES];
  agg[i] = rn * rn * h[i];
}

// agg[dst] += w * h[src]   (10 fp32 hardware atomics per edge; h + edges L2-resident)
__global__ void scatter_kernel(const int* __restrict__ src, const int* __restrict__ dst,
                               const float* __restrict__ ew,
                               const float* __restrict__ h, float* agg) {
  int e = blockIdx.x * blockDim.x + threadIdx.x;
  if (e >= NEDGES) return;
  int s = src[e], d = dst[e];
  float w = ew[e];
  const float2* hp = (const float2*)(h + (size_t)s * CLASSES);  // 8B aligned (40*s)
  float* ap = agg + (size_t)d * CLASSES;
#pragma unroll
  for (int k = 0; k < 5; ++k) {
    float2 v = hp[k];
    unsafeAtomicAdd(&ap[2 * k],     w * v.x);
    unsafeAtomicAdd(&ap[2 * k + 1], w * v.y);
  }
}

// out = ALPHA*agg + (1-ALPHA)*h0
__global__ void combine_kernel(const float* agg, const float* h0, float* out) {
  int i = blockIdx.x * blockDim.x + threadIdx.x;
  if (i >= NNODES * CLASSES) return;
  out[i] = ALPHA * agg[i] + (1.0f - ALPHA) * h0[i];
}

// ---------------- WMMA MLP: h0 = relu(xb@w1b.T+b1)@w2b.T + b2 ----------------

#define MLP_WPB 4   // waves (16-row tiles) per block

__global__ void mlp_kernel(const __bf16* __restrict__ xb, const __bf16* __restrict__ w1b,
                           const float* __restrict__ b1,  const __bf16* __restrict__ w2b,
                           const float* __restrict__ b2,  float* __restrict__ h0) {
  __shared__ __bf16 lds[MLP_WPB][16][HIDDEN];          // 8 KB, bf16 hidden tiles
  int wave = threadIdx.x >> 5;
  int lane = threadIdx.x & 31;
  int tile = blockIdx.x * MLP_WPB + wave;
  bool valid = tile < NTILES;
  int t2 = valid ? tile : (NTILES - 1);                // clamp for safe loads
  int row0 = t2 * 16;
  int lr = lane & 15;
  int hi = (lane >> 4) << 3;

  // layer 1: [16x128] @ [128x64] with 4 K-steps, 4 hidden tiles
  bf16x16 afrag[4];
#pragma unroll
  for (int ks = 0; ks < 4; ++ks)
    afrag[ks] = load_frag(xb, row0 + lr, FEATS, ks * 32, lane);

#pragma unroll
  for (int ht = 0; ht < 4; ++ht) {
    f32x8 acc = {};
#pragma unroll
    for (int ks = 0; ks < 4; ++ks) {
      bf16x16 bfrag = load_frag(w1b, ht * 16 + lr, FEATS, ks * 32, lane);
      acc = wmma_bf16(afrag[ks], bfrag, acc);
    }
    int col = ht * 16 + lr;
    float bias = b1[col];
#pragma unroll
    for (int g = 0; g < 8; ++g) {
      int m = g + hi;                                   // D layout: M=g (+8 for hi lanes)
      lds[wave][m][col] = f2bf(fmaxf(acc[g] + bias, 0.0f));
    }
  }
  __syncthreads();

  // layer 2: [16x64] @ [64x16] (w2b rows 10..15 are zero; cols >=10 discarded)
  f32x8 acc2 = {};
#pragma unroll
  for (int ks = 0; ks < 2; ++ks) {
    bf16x16 a2 = load_frag(&lds[wave][0][0], lr, HIDDEN, ks * 32, lane);
    bf16x16 bf = load_frag(w2b, lr, HIDDEN, ks * 32, lane);
    acc2 = wmma_bf16(a2, bf, acc2);
  }
  if (valid && lr < CLASSES) {
    float bias = b2[lr];
#pragma unroll
    for (int g = 0; g < 8; ++g) {
      int m = g + hi;
      h0[(size_t)(row0 + m) * CLASSES + lr] = acc2[g] + bias;
    }
  }
}

// ---------------- WMMA xa precompute: xa = xb @ wab.T + ba1 ----------------

__global__ void xa_kernel(const __bf16* __restrict__ xb, const __bf16* __restrict__ wab,
                          const float* __restrict__ ba1, float* __restrict__ xa) {
  int wave = threadIdx.x >> 5;
  int lane = threadIdx.x & 31;
  int tile = blockIdx.x * MLP_WPB + wave;
  if (tile >= NTILES) return;
  int row0 = tile * 16;
  int lr = lane & 15;
  int hi = (lane >> 4) << 3;

  bf16x16 afrag[4];
#pragma unroll
  for (int ks = 0; ks < 4; ++ks)
    afrag[ks] = load_frag(xb, row0 + lr, FEATS, ks * 32, lane);

  // 9 output tiles cover 144 cols (rows >=131 of wab are zero-padded)
#pragma unroll
  for (int jt = 0; jt < 9; ++jt) {
    int j = jt * 16 + lr;
    f32x8 acc = {};
#pragma unroll
    for (int ks = 0; ks < 4; ++ks) {
      bf16x16 bfrag = load_frag(wab, j, FEATS, ks * 32, lane);
      acc = wmma_bf16(afrag[ks], bfrag, acc);
    }
    float bias = (j < ADJ_HID) ? ba1[j] : 0.0f;
#pragma unroll
    for (int g = 0; g < 8; ++g) {
      int m = g + hi;
      xa[(size_t)(row0 + m) * XA_STRIDE + j] = acc[g] + bias;
    }
  }
}

// ---------------- adjustment MLP (restructured), wave per node ----------------
// hadj[n,c] = ba2 + sum_j wa2[j] * relu( xa[n,j] + h[n,c]*wa1[j,0] + wa1[j,1+c] )

__global__ void adjust_kernel(const float* __restrict__ xa, const float* __restrict__ h,
                              const float* __restrict__ wa1, const float* __restrict__ wa2,
                              const float* __restrict__ ba2, float* __restrict__ out) {
  int gid  = blockIdx.x * blockDim.x + threadIdx.x;
  int n    = gid >> 5;
  int lane = gid & 31;
  if (n >= NNODES) return;

  float xaj[5], c0[5], w2j[5];
  int jj[5];
#pragma unroll
  for (int t = 0; t < 5; ++t) {
    int j = lane + 32 * t;
    bool ok = j < ADJ_HID;
    jj[t]  = ok ? j : 0;
    xaj[t] = ok ? xa[(size_t)n * XA_STRIDE + j] : 0.0f;
    c0[t]  = ok ? wa1[(size_t)j * 139] : 0.0f;
    w2j[t] = ok ? wa2[j] : 0.0f;                       // zero weight kills padded lanes
  }
  float bias2 = ba2[0];
#pragma unroll
  for (int c = 0; c < CLASSES; ++c) {
    float hv = h[(size_t)n * CLASSES + c];
    float acc = 0.0f;
#pragma unroll
    for (int t = 0; t < 5; ++t) {
      float ci  = wa1[(size_t)jj[t] * 139 + 1 + c];    // weights: L1/L2 resident
      float pre = xaj[t] + hv * c0[t] + ci;
      acc += w2j[t] * fmaxf(pre, 0.0f);
    }
#pragma unroll
    for (int off = 16; off > 0; off >>= 1)
      acc += __shfl_xor(acc, off, 32);
    if (lane == 0) out[(size_t)n * CLASSES + c] = acc + bias2;
  }
}

// ---------------- host-side launch ----------------

static inline size_t alignup(size_t v) { return (v + 255) & ~(size_t)255; }

extern "C" void kernel_launch(void* const* d_in, const int* in_sizes, int n_in,
                              void* d_out, int out_size, void* d_ws, size_t ws_size,
                              hipStream_t stream) {
  const float* x   = (const float*)d_in[0];
  const float* w1  = (const float*)d_in[1];
  const float* b1  = (const float*)d_in[2];
  const float* w2  = (const float*)d_in[3];
  const float* b2  = (const float*)d_in[4];
  const float* wa1 = (const float*)d_in[5];
  const float* ba1 = (const float*)d_in[6];
  const float* wa2 = (const float*)d_in[7];
  const float* ba2 = (const float*)d_in[8];
  const long long* edges = (const long long*)d_in[9];
  float* out = (float*)d_out;

  // workspace carve-up
  char* p = (char*)d_ws;
  size_t o = 0;
  const size_t hcbytes = (size_t)NNODES * CLASSES * sizeof(float);
  float* h0   = (float*)(p + o); o += alignup(hcbytes);
  float* h    = (float*)(p + o); o += alignup(hcbytes);
  float* agg  = (float*)(p + o); o += alignup(hcbytes);
  float* h2_0 = (float*)(p + o); o += alignup(hcbytes);
  float* r    = (float*)(p + o); o += alignup((size_t)NNODES * sizeof(float));
  float* deg  = (float*)(p + o); o += alignup((size_t)NNODES * sizeof(float));
  int*   src32 = (int*)(p + o); o += alignup((size_t)NEDGES * sizeof(int));
  int*   dst32 = (int*)(p + o); o += alignup((size_t)NEDGES * sizeof(int));
  float* ew   = (float*)(p + o); o += alignup((size_t)NEDGES * sizeof(float));
  float* xa   = (float*)(p + o); o += alignup((size_t)NNODES * XA_STRIDE * sizeof(float));
  __bf16* xb  = (__bf16*)(p + o); o += alignup((size_t)NNODES * FEATS * sizeof(__bf16));
  __bf16* w1b = (__bf16*)(p + o); o += alignup((size_t)HIDDEN * FEATS * sizeof(__bf16));
  __bf16* w2b = (__bf16*)(p + o); o += alignup((size_t)16 * HIDDEN * sizeof(__bf16));
  __bf16* wab = (__bf16*)(p + o); o += alignup((size_t)XA_STRIDE * FEATS * sizeof(__bf16));
  (void)ws_size; (void)n_in; (void)in_sizes; (void)out_size;

  const int T = 256;
  dim3 bN((NNODES + T - 1) / T);
  dim3 bE((NEDGES + T - 1) / T);
  dim3 bNC((NNODES * CLASSES + T - 1) / T);
  dim3 bX((NNODES * FEATS + T - 1) / T);
  dim3 bW((XA_STRIDE * FEATS + T - 1) / T);
  dim3 bMLP((NTILES + MLP_WPB - 1) / MLP_WPB);
  dim3 bADJ(NNODES / 8);                 // 8 waves (nodes) per 256-thread block

  // bf16 repack prepass
  cvt_x_kernel<<<bX, T, 0, stream>>>(x, xb);
  cvt_w_kernel<<<bW, T, 0, stream>>>(w1, w2, wa1, w1b, w2b, wab);

  // graph prep
  zero_deg_kernel<<<bN, T, 0, stream>>>(deg);
  count_deg_kernel<<<bE, T, 0, stream>>>(edges, deg);
  compute_r_kernel<<<bN, T, 0, stream>>>(deg, r);
  edge_prep_kernel<<<bE, T, 0, stream>>>(edges, r, src32, dst32, ew);

  // MLP stage (WMMA) -> h0
  mlp_kernel<<<bMLP, 32 * MLP_WPB, 0, stream>>>(xb, w1b, b1, w2b, b2, h0);

  // diffusion 1: h starts at h0
  for (int it = 0; it < DEPTH; ++it) {
    const float* hs = (it == 0) ? h0 : h;
    init_agg_kernel<<<bNC, T, 0, stream>>>(r, hs, agg);
    scatter_kernel<<<bE, T, 0, stream>>>(src32, dst32, ew, hs, agg);
    combine_kernel<<<bNC, T, 0, stream>>>(agg, h0, h);
  }

  // adjustment MLP: xa precompute (WMMA) + wave-per-node fused stage
  xa_kernel<<<bMLP, 32 * MLP_WPB, 0, stream>>>(xb, wab, ba1, xa);
  adjust_kernel<<<bADJ, T, 0, stream>>>(xa, h, wa1, wa2, ba2, h2_0);

  // diffusion 2: h starts at h2_0, final iteration writes d_out
  for (int it = 0; it < DEPTH; ++it) {
    const float* hs = (it == 0) ? h2_0 : h;
    float* hd = (it == DEPTH - 1) ? out : h;
    init_agg_kernel<<<bNC, T, 0, stream>>>(r, hs, agg);
    scatter_kernel<<<bE, T, 0, stream>>>(src32, dst32, ew, hs, agg);
    combine_kernel<<<bNC, T, 0, stream>>>(agg, h2_0, hd);
  }
}
